// CrossYearEpisodicMemory_51402168599306
// MI455X (gfx1250) — compile-verified, matrix operand order checked by reading, not run on hardware
//
#include <hip/hip_runtime.h>
#include <hip/hip_bf16.h>
#include <math.h>

// Problem constants
#define Bq   16
#define Tt   96
#define Nn   256
#define Dd   128
#define Mm   1196
#define MP   1200      // padded memory rows (multiple of 16)
#define Kk   8
#define Hh   4
#define HDd  32
#define TOUT 97
#define TP   112       // padded t (multiple of 16)
#define ND   32768     // N*D
#define KSPLIT 32      // K-partitions for sim GEMM

typedef __attribute__((ext_vector_type(16))) __bf16 v16bf;
typedef __attribute__((ext_vector_type(8)))  float  v8f;
typedef __attribute__((ext_vector_type(8)))  unsigned short ushort8;
typedef __attribute__((ext_vector_type(4)))  unsigned short us4;

union TileU { v16bf bf; ushort8 h[2]; };

// ---- helpers -------------------------------------------------------------

__device__ __forceinline__ unsigned short f2bf(float f) {
    unsigned int x = __float_as_uint(f);
    unsigned int r = x + 0x7fffu + ((x >> 16) & 1u);   // round-to-nearest-even
    return (unsigned short)(r >> 16);
}

__device__ __forceinline__ float gelu_exact(float x) {
    return 0.5f * x * (1.0f + erff(x * 0.70710678118654752f));
}

// Load a 16-lane-striped WMMA operand (A row-block or B col-block) from a
// K-contiguous row-major bf16 matrix.  Per ISA 16-bit layout: lane l
// (g=l>>4) holds K = kc+8g+[0..7] and kc+16+8g+[0..7].
__device__ __forceinline__ v16bf load_tile(const unsigned short* base, int row,
                                           int ld, int kc) {
    const int l = threadIdx.x & 31;
    const unsigned short* p = base + (size_t)row * ld + kc + ((l >> 4) << 3);
    TileU u;
    u.h[0] = *(const ushort8*)(p);
    u.h[1] = *(const ushort8*)(p + 16);
    return u.bf;
}

__device__ __forceinline__ v8f wmma_bf16(v16bf a, v16bf b, v8f c) {
    // (neg_a, A, neg_b, B, c_mod, C, reuse_a, reuse_b)
    return __builtin_amdgcn_wmma_f32_16x16x32_bf16(false, a, false, b,
                                                   (short)0, c, false, false);
}

__device__ __forceinline__ float wave_sum32(float v) {
    for (int o = 16; o > 0; o >>= 1) v += __shfl_xor(v, o, 32);
    return v;
}

// ---- kernels -------------------------------------------------------------

// fp32 -> bf16, vectorized (n must be a multiple of 4)
__global__ void k_convert(const float* __restrict__ src,
                          unsigned short* __restrict__ dst, int n4) {
    int i = blockIdx.x * blockDim.x + threadIdx.x;
    if (i < n4) {
        float4 f = ((const float4*)src)[i];
        us4 o = { f2bf(f.x), f2bf(f.y), f2bf(f.z), f2bf(f.w) };
        ((us4*)dst)[i] = o;
    }
}

// Convert memory bank to bf16 (padded to MP rows) + per-row L2 norm
__global__ void k_bank_convert(const float* __restrict__ bank,
                               unsigned short* __restrict__ bankbf,
                               float* __restrict__ memnorm) {
    __shared__ float red[8];
    const int m = blockIdx.x;
    const int tid = threadIdx.x;
    float ss = 0.f;
    us4* d4 = (us4*)(bankbf + (size_t)m * ND);
    if (m < Mm) {
        const float4* s4 = (const float4*)(bank + (size_t)m * ND);
        for (int i = tid; i < ND / 4; i += 256) {
            float4 f = s4[i];
            us4 o = { f2bf(f.x), f2bf(f.y), f2bf(f.z), f2bf(f.w) };
            d4[i] = o;
            ss += f.x * f.x + f.y * f.y + f.z * f.z + f.w * f.w;
        }
    } else {
        for (int i = tid; i < ND / 4; i += 256) { us4 z = {0,0,0,0}; d4[i] = z; }
    }
    ss = wave_sum32(ss);
    if ((tid & 31) == 0) red[tid >> 5] = ss;
    __syncthreads();
    if (tid == 0) {
        float t = 0.f;
        for (int w = 0; w < 8; ++w) t += red[w];
        memnorm[m] = (m < Mm) ? sqrtf(t) : 1.0f;
    }
}

// Depthwise conv (kernel 12, pad 6/6) + GELU  -> xct[b][t][c] bf16 (t padded)
__global__ void k_dwconv(const float* __restrict__ x,
                         const float* __restrict__ dw_w,
                         const float* __restrict__ dw_b,
                         unsigned short* __restrict__ xct) {
    __shared__ float xs[Tt];
    __shared__ float wv[12];
    const int bn = blockIdx.x;           // b*256+n
    const int b = bn >> 8, n = bn & 255;
    const int tid = threadIdx.x;
    if (tid < Tt) xs[tid] = x[((size_t)b * Tt + tid) * Nn + n];
    if (tid < 12) wv[tid] = dw_w[n * 12 + tid];
    __syncthreads();
    if (tid < TP) {
        float val = 0.f;
        if (tid < TOUT) {
            float acc = dw_b[n];
            #pragma unroll
            for (int j = 0; j < 12; ++j) {
                int tt = tid + j - 6;
                if (tt >= 0 && tt < Tt) acc += xs[tt] * wv[j];
            }
            val = gelu_exact(acc);
        }
        xct[((size_t)b * TP + tid) * Nn + n] = f2bf(val);
    }
}

// Issue 4 async b128 loads copying a contiguous 8KB tile -> LDS (128 threads)
__device__ __forceinline__ void stage_async_8k(const unsigned short* gsrc,
                                               unsigned lds_base, int tid) {
    unsigned long long g = (unsigned long long)gsrc + (unsigned)tid * 16u;
    unsigned lds = lds_base + (unsigned)tid * 16u;
    #pragma unroll
    for (int i = 0; i < 4; ++i) {
        asm volatile("global_load_async_to_lds_b128 %0, %1, off"
                     :: "v"(lds), "v"(g) : "memory");
        g += 2048u; lds += 2048u;
    }
}

// Pointwise GEMM: y[o,t] = gelu(pw_w[o,:] . xc[:,t] + pw_b[o]), mean over t<97
// 4 waves per block (4 o-tiles); the shared 16x256 xct t-tile is staged into
// LDS with async-to-LDS DMA, double-buffered against the 8-WMMA chain.
__global__ void k_pointwise(const unsigned short* __restrict__ pwbf,
                            const unsigned short* __restrict__ xct,
                            const float* __restrict__ pw_b,
                            float* __restrict__ xm) {
    __shared__ __align__(16) unsigned short sb[2][16 * Nn];  // 2 x 8KB
    const int b = blockIdx.y;            // 0..15
    const int tid = threadIdx.x;         // 0..127
    const int wid = tid >> 5;
    const int l = tid & 31;
    const int g = l >> 4;
    const int otile = blockIdx.x * 4 + wid;      // 0..2047
    const unsigned short* xb = xct + (size_t)b * TP * Nn;
    const unsigned lds0 = (unsigned)(unsigned long long)(void*)&sb[0][0];
    const unsigned lds1 = (unsigned)(unsigned long long)(void*)&sb[1][0];

    v16bf amat[8];
    #pragma unroll
    for (int kk = 0; kk < 8; ++kk)
        amat[kk] = load_tile(pwbf, otile * 16 + (l & 15), Nn, kk * 32);

    float bias[8], msum[8];
    #pragma unroll
    for (int r = 0; r < 8; ++r) {
        bias[r] = pw_b[otile * 16 + r + 8 * g];
        msum[r] = 0.f;
    }

    // prime tile 0
    stage_async_8k(xb, lds0, tid);

    for (int tt = 0; tt < 7; ++tt) {
        const int cb = tt & 1;
        if (tt < 6) {
            stage_async_8k(xb + (size_t)(tt + 1) * 16 * Nn,
                           (tt & 1) ? lds0 : lds1, tid);
            asm volatile("s_wait_asynccnt 4" ::: "memory");
        } else {
            asm volatile("s_wait_asynccnt 0" ::: "memory");
        }
        __syncthreads();                 // staged tile visible to all waves

        v8f acc;
        #pragma unroll
        for (int i = 0; i < 8; ++i) acc[i] = 0.f;
        #pragma unroll
        for (int kk = 0; kk < 8; ++kk) {
            TileU u;
            const unsigned short* p =
                &sb[cb][(l & 15) * Nn + kk * 32 + (g << 3)];
            u.h[0] = *(const ushort8*)(p);
            u.h[1] = *(const ushort8*)(p + 16);
            acc = wmma_bf16(amat[kk], u.bf, acc);
        }

        const int t = tt * 16 + (l & 15);
        if (t < TOUT) {
            #pragma unroll
            for (int r = 0; r < 8; ++r)
                msum[r] += gelu_exact(acc[r] + bias[r]);
        }
        __syncthreads();                 // readers done before next overwrite
    }

    #pragma unroll
    for (int r = 0; r < 8; ++r) {
        for (int o = 1; o < 16; o <<= 1) msum[r] += __shfl_xor(msum[r], o, 16);
        if ((l & 15) == 0)
            xm[(size_t)b * ND + otile * 16 + r + 8 * g] = msum[r] / (float)TOUT;
    }
}

// LayerNorm over D=128 per (b,n); writes q (f32, second output), qbf, qnorm2.
__global__ void k_layernorm(const float* __restrict__ xm,
                            const float* __restrict__ ln_g,
                            const float* __restrict__ ln_b,
                            float* __restrict__ q_out,
                            unsigned short* __restrict__ qbf,
                            float* __restrict__ qnorm2) {
    __shared__ float red[4];
    const int bn = blockIdx.x;
    const int dd = threadIdx.x;
    const int w = dd >> 5;
    float x = xm[(size_t)bn * Dd + dd];

    float s = wave_sum32(x);
    if ((dd & 31) == 0) red[w] = s;
    __syncthreads();
    float mu = (red[0] + red[1] + red[2] + red[3]) / (float)Dd;
    __syncthreads();

    float dv = x - mu;
    float s2 = wave_sum32(dv * dv);
    if ((dd & 31) == 0) red[w] = s2;
    __syncthreads();
    float var = (red[0] + red[1] + red[2] + red[3]) / (float)Dd;
    __syncthreads();

    float y = dv * rsqrtf(var + 1e-5f) * ln_g[dd] + ln_b[dd];
    q_out[(size_t)bn * Dd + dd] = y;
    qbf[(size_t)bn * Dd + dd] = f2bf(y);

    float sq = wave_sum32(y * y);
    if ((dd & 31) == 0) red[w] = sq;
    __syncthreads();
    if (dd == 0)
        atomicAdd(&qnorm2[bn >> 8], red[0] + red[1] + red[2] + red[3]);
}

// sim GEMM: simraw[b][m] += q_b . mem_m, K split across gridDim.y.
// Software-pipelined + explicit prefetch on the HBM bank stream.
__global__ void k_simgemm(const unsigned short* __restrict__ qbf,
                          const unsigned short* __restrict__ bankbf,
                          float* __restrict__ simraw) {
    const int mtile = blockIdx.x;        // 0..74
    const int kz = blockIdx.y;           // 0..KSPLIT-1
    const int l = threadIdx.x;
    const int mrow = mtile * 16 + (l & 15);
    const int arow = (l & 15);
    const int iters = (ND / KSPLIT) / 32;
    const int k0 = kz * (ND / KSPLIT);

    v8f acc;
    #pragma unroll
    for (int i = 0; i < 8; ++i) acc[i] = 0.f;

    v16bf a = load_tile(qbf, arow, ND, k0);
    v16bf bm = load_tile(bankbf, mrow, ND, k0);
    for (int kk = 1; kk < iters; ++kk) {
        const int kc = k0 + kk * 32;
        // prefetch the bank stream ~2KB ahead (speculative, lane-private)
        __builtin_prefetch(bankbf + (size_t)mrow * ND + kc + 1024, 0, 0);
        v16bf an = load_tile(qbf, arow, ND, kc);
        v16bf bn = load_tile(bankbf, mrow, ND, kc);
        acc = wmma_bf16(a, bm, acc);
        a = an; bm = bn;
    }
    acc = wmma_bf16(a, bm, acc);

    #pragma unroll
    for (int r = 0; r < 8; ++r) {
        const int bb = r + 8 * (l >> 4);
        atomicAdd(&simraw[bb * MP + mrow], acc[r]);
    }
}

// Finalize sim (normalize, season mask, diversity), then top-8 per batch.
__global__ void k_topk(const float* __restrict__ simraw,
                       const float* __restrict__ qnorm2,
                       const float* __restrict__ memnorm,
                       const int* __restrict__ season_q,
                       const int* __restrict__ mem_seasons,
                       const float* __restrict__ year_q,
                       const float* __restrict__ mem_years,
                       int* __restrict__ idxb) {
    __shared__ float sv[MP];
    __shared__ float lv[256];
    __shared__ int   li[256];
    const int b = blockIdx.x;
    const int tid = threadIdx.x;
    const float qn = fmaxf(sqrtf(qnorm2[b]), 1e-12f);
    const int sq = season_q[b];
    const float yq = year_q[b];
    for (int m = tid; m < MP; m += 256) {
        float s = -3.0e38f;
        if (m < Mm) {
            s = simraw[b * MP + m] / (qn * fmaxf(memnorm[m], 1e-12f));
            if (sq != mem_seasons[m]) s = -10000.0f;
            float dy = fabsf(yq - mem_years[m]);
            float div = 1.0f - expf(-dy / 2.0f);
            s = s * (0.5f + 0.5f * div);
        }
        sv[m] = s;
    }
    __syncthreads();
    for (int kpass = 0; kpass < Kk; ++kpass) {
        float bv = -3.4e38f; int bi = MP;
        for (int m = tid; m < MP; m += 256) {
            float v = sv[m];
            if (v > bv || (v == bv && m < bi)) { bv = v; bi = m; }
        }
        lv[tid] = bv; li[tid] = bi;
        __syncthreads();
        for (int s = 128; s > 0; s >>= 1) {
            if (tid < s) {
                if (lv[tid + s] > lv[tid] ||
                    (lv[tid + s] == lv[tid] && li[tid + s] < li[tid])) {
                    lv[tid] = lv[tid + s]; li[tid] = li[tid + s];
                }
            }
            __syncthreads();
        }
        if (tid == 0) { idxb[b * Kk + kpass] = li[0]; sv[li[0]] = -3.4e38f; }
        __syncthreads();
    }
}

// Gather kv rows (bf16) via CDNA5 async LDS path:
// global -> LDS (global_load_async_to_lds_b64) -> global
// (global_store_async_from_lds_b64), tracked with ASYNCcnt.
__global__ void k_gather(const unsigned short* __restrict__ bankbf,
                         const int* __restrict__ idxb,
                         unsigned short* __restrict__ kvbf) {
    __shared__ unsigned char sbuf[8 * 256];       // 256B per wave
    const int wid = threadIdx.x >> 5;
    const int l = threadIdx.x & 31;
    const int row = blockIdx.x * 8 + wid;         // 0..32767
    const int k = row & 7;
    const int n = (row >> 3) & 255;
    const int b = row >> 11;
    const int mem = idxb[b * Kk + k];

    unsigned long long gsrc =
        (unsigned long long)(bankbf + ((size_t)mem * Nn + n) * Dd) + (unsigned)l * 8u;
    unsigned long long gdst =
        (unsigned long long)(kvbf + (size_t)row * Dd) + (unsigned)l * 8u;
    // flat LDS addresses: low 32 bits of the generic pointer
    unsigned lds = (unsigned)(unsigned long long)(void*)&sbuf[wid * 256 + l * 8];

    asm volatile("global_load_async_to_lds_b64 %0, %1, off"
                 :: "v"(lds), "v"(gsrc) : "memory");
    asm volatile("s_wait_asynccnt 0" ::: "memory");
    asm volatile("global_store_async_from_lds_b64 %0, %1, off"
                 :: "v"(gdst), "v"(lds) : "memory");
    asm volatile("s_wait_asynccnt 0" ::: "memory");
}

// Generic GEMM: out[r][c] = A[r,:] . W[c,:] + bias[c]   (K-contig bf16 operands)
__global__ void k_gemm(const unsigned short* __restrict__ A, int lda,
                       const unsigned short* __restrict__ W, int ldw,
                       const float* __restrict__ bias,
                       float* __restrict__ outf,
                       unsigned short* __restrict__ outb,
                       int ldo, int kdim) {
    const int l = threadIdx.x;
    const int rowA = blockIdx.x * 16 + (l & 15);
    const int colW = blockIdx.y * 16 + (l & 15);
    v8f acc;
    #pragma unroll
    for (int i = 0; i < 8; ++i) acc[i] = 0.f;

    v16bf a = load_tile(A, rowA, lda, 0);
    v16bf w = load_tile(W, colW, ldw, 0);
    for (int kc = 32; kc < kdim; kc += 32) {
        v16bf an = load_tile(A, rowA, lda, kc);
        v16bf wn = load_tile(W, colW, ldw, kc);
        acc = wmma_bf16(a, w, acc);
        a = an; w = wn;
    }
    acc = wmma_bf16(a, w, acc);

    const int col = blockIdx.y * 16 + (l & 15);
    const float bc = bias ? bias[col] : 0.f;
    #pragma unroll
    for (int r = 0; r < 8; ++r) {
        const int m = blockIdx.x * 16 + r + 8 * (l >> 4);
        float v = acc[r] + bc;
        if (outf) outf[(size_t)m * ldo + col] = v;
        if (outb) outb[(size_t)m * ldo + col] = f2bf(v);
    }
}

// Per-(b,n) attention over K=8 retrieved rows; one wave per head.
__global__ void k_attn(const float* __restrict__ qp,
                       const float* __restrict__ kp,
                       const float* __restrict__ vp,
                       unsigned short* __restrict__ ctxbf) {
    const int row = blockIdx.x;          // b*256+n
    const int h = threadIdx.x >> 5;
    const int e = threadIdx.x & 31;
    const float qv = qp[(size_t)row * Dd + h * HDd + e];
    float s[Kk];
    #pragma unroll
    for (int k = 0; k < Kk; ++k) {
        float p = qv * kp[((size_t)(row * Kk + k)) * Dd + h * HDd + e];
        p = wave_sum32(p);
        s[k] = p * 0.17677669529663687f;   // 1/sqrt(32)
    }
    float mx = s[0];
    #pragma unroll
    for (int k = 1; k < Kk; ++k) mx = fmaxf(mx, s[k]);
    float sum = 0.f;
    #pragma unroll
    for (int k = 0; k < Kk; ++k) { s[k] = expf(s[k] - mx); sum += s[k]; }
    const float inv = 1.0f / sum;
    float ctx = 0.f;
    #pragma unroll
    for (int k = 0; k < Kk; ++k)
        ctx += s[k] * inv * vp[((size_t)(row * Kk + k)) * Dd + h * HDd + e];
    ctxbf[(size_t)row * Dd + h * HDd + e] = f2bf(ctx);
}

// ---- launch --------------------------------------------------------------

extern "C" void kernel_launch(void* const* d_in, const int* in_sizes, int n_in,
                              void* d_out, int out_size, void* d_ws, size_t ws_size,
                              hipStream_t stream) {
    (void)in_sizes; (void)n_in; (void)out_size; (void)ws_size;
    const float* x_scalar   = (const float*)d_in[0];
    const int*   season_q   = (const int*)  d_in[1];
    const float* year_q     = (const float*)d_in[2];
    const float* dw_w       = (const float*)d_in[3];
    const float* dw_b       = (const float*)d_in[4];
    const float* pw_w       = (const float*)d_in[5];
    const float* pw_b       = (const float*)d_in[6];
    const float* ln_g       = (const float*)d_in[7];
    const float* ln_b       = (const float*)d_in[8];
    const float* bank       = (const float*)d_in[9];
    const int*   mem_seas   = (const int*)  d_in[10];
    const float* mem_years  = (const float*)d_in[11];
    const float* in_proj_w  = (const float*)d_in[12];
    const float* in_proj_b  = (const float*)d_in[13];
    const float* out_proj_w = (const float*)d_in[14];
    const float* out_proj_b = (const float*)d_in[15];
    const float* proj_w     = (const float*)d_in[16];
    const float* proj_b     = (const float*)d_in[17];

    float* out_final = (float*)d_out;                    // (B,N,D)
    float* q_out = (float*)d_out + (size_t)Bq * Nn * Dd; // (B,N,D)

    // workspace carve (256B aligned)
    size_t cur = 0;
    char* base = (char*)d_ws;
    auto carve = [&](size_t bytes) -> void* {
        cur = (cur + 255) & ~(size_t)255;
        void* p = base + cur;
        cur += bytes;
        return p;
    };
    unsigned short* bankbf    = (unsigned short*)carve((size_t)MP * ND * 2);
    float*          memnorm   = (float*)         carve((size_t)MP * 4);
    unsigned short* pwbf      = (unsigned short*)carve((size_t)ND * Nn * 2);
    unsigned short* inprojbf  = (unsigned short*)carve((size_t)3 * Dd * Dd * 2);
    unsigned short* outprojbf = (unsigned short*)carve((size_t)Dd * Dd * 2);
    unsigned short* projbf    = (unsigned short*)carve((size_t)Dd * Dd * 2);
    unsigned short* xct       = (unsigned short*)carve((size_t)Bq * TP * Nn * 2);
    float*          xm        = (float*)         carve((size_t)Bq * ND * 4);
    unsigned short* qbf       = (unsigned short*)carve((size_t)Bq * ND * 2);
    float*          qnorm2    = (float*)         carve((size_t)Bq * 4);
    float*          simraw    = (float*)         carve((size_t)Bq * MP * 4);
    int*            idxb      = (int*)           carve((size_t)Bq * Kk * 4);
    unsigned short* kvbf      = (unsigned short*)carve((size_t)Bq * Nn * Kk * Dd * 2);
    float*          qp        = (float*)         carve((size_t)Bq * Nn * Dd * 4);
    float*          kp        = (float*)         carve((size_t)Bq * Nn * Kk * Dd * 4);
    float*          vp        = (float*)         carve((size_t)Bq * Nn * Kk * Dd * 4);
    unsigned short* ctxbf     = (unsigned short*)carve((size_t)Bq * Nn * Dd * 2);
    unsigned short* out1bf    = (unsigned short*)carve((size_t)Bq * Nn * Dd * 2);

    hipMemsetAsync(qnorm2, 0, (size_t)Bq * 4, stream);
    hipMemsetAsync(simraw, 0, (size_t)Bq * MP * 4, stream);

    // weight conversions fp32->bf16 (vectorized x4)
    k_convert<<<(ND * Nn / 4 + 255) / 256, 256, 0, stream>>>(pw_w, pwbf, ND * Nn / 4);
    k_convert<<<(3 * Dd * Dd / 4 + 255) / 256, 256, 0, stream>>>(in_proj_w, inprojbf, 3 * Dd * Dd / 4);
    k_convert<<<(Dd * Dd / 4 + 255) / 256, 256, 0, stream>>>(out_proj_w, outprojbf, Dd * Dd / 4);
    k_convert<<<(Dd * Dd / 4 + 255) / 256, 256, 0, stream>>>(proj_w, projbf, Dd * Dd / 4);

    k_bank_convert<<<MP, 256, 0, stream>>>(bank, bankbf, memnorm);
    k_dwconv<<<Bq * Nn, 128, 0, stream>>>(x_scalar, dw_w, dw_b, xct);
    k_pointwise<<<dim3(ND / 16 / 4, Bq), 128, 0, stream>>>(pwbf, xct, pw_b, xm);
    k_layernorm<<<Bq * Nn, 128, 0, stream>>>(xm, ln_g, ln_b, q_out, qbf, qnorm2);
    k_simgemm<<<dim3(MP / 16, KSPLIT), 32, 0, stream>>>(qbf, bankbf, simraw);
    k_topk<<<Bq, 256, 0, stream>>>(simraw, qnorm2, memnorm, season_q, mem_seas,
                                   year_q, mem_years, idxb);
    k_gather<<<Bq * Nn * Kk / 8, 256, 0, stream>>>(bankbf, idxb, kvbf);

    // projections: qp, kp, vp
    k_gemm<<<dim3(Bq * Nn / 16, Dd / 16), 32, 0, stream>>>(
        qbf, Dd, inprojbf, Dd, in_proj_b, qp, nullptr, Dd, Dd);
    k_gemm<<<dim3(Bq * Nn * Kk / 16, Dd / 16), 32, 0, stream>>>(
        kvbf, Dd, inprojbf + Dd * Dd, Dd, in_proj_b + Dd, kp, nullptr, Dd, Dd);
    k_gemm<<<dim3(Bq * Nn * Kk / 16, Dd / 16), 32, 0, stream>>>(
        kvbf, Dd, inprojbf + 2 * Dd * Dd, Dd, in_proj_b + 2 * Dd, vp, nullptr, Dd, Dd);

    k_attn<<<Bq * Nn, 128, 0, stream>>>(qp, kp, vp, ctxbf);

    // out = (ctx @ Wout^T + bout) @ Wproj^T + bproj
    k_gemm<<<dim3(Bq * Nn / 16, Dd / 16), 32, 0, stream>>>(
        ctxbf, Dd, outprojbf, Dd, out_proj_b, nullptr, out1bf, Dd, Dd);
    k_gemm<<<dim3(Bq * Nn / 16, Dd / 16), 32, 0, stream>>>(
        out1bf, Dd, projbf, Dd, proj_b, out_final, nullptr, Dd, Dd);
}
